// GELU18_17566416240700
// MI455X (gfx1250) — matrix operand order repaired and checked by reading, not compiled, not run on hardware
//
#include <hip/hip_runtime.h>
#include <hip/hip_bf16.h>
#include <math.h>

// Problem dims (fixed by the reference)
#define BB 8
#define TT 2048
#define DD 512
#define NT (TT / 16)   // 128 row/col tiles per batch

typedef _Float16 h8   __attribute__((ext_vector_type(8)));
typedef _Float16 v16h __attribute__((ext_vector_type(16)));
typedef float    v8f  __attribute__((ext_vector_type(8)));
typedef float    v4f  __attribute__((ext_vector_type(4)));

// ---------------------------------------------------------------------------
// Kernel 1: per-row L2 normalize (clamped at 1e-12) and convert to f16.
// One wave (32 lanes) per row; each lane owns 16 consecutive floats.
// ---------------------------------------------------------------------------
__global__ __launch_bounds__(256) void normalize_f16_kernel(
    const float* __restrict__ x, _Float16* __restrict__ xn) {
  const int lane = threadIdx.x & 31;
  const int wave = threadIdx.x >> 5;
  const size_t row = (size_t)blockIdx.x * 8 + wave;   // B*T rows total

  const float* px = x + row * DD + lane * 16;
  v4f t[4];
  float acc = 0.0f;
#pragma unroll
  for (int i = 0; i < 4; ++i) {
    t[i] = ((const v4f*)px)[i];
#pragma unroll
    for (int j = 0; j < 4; ++j) acc += t[i][j] * t[i][j];
  }
  // wave32 tree reduction
#pragma unroll
  for (int m = 16; m >= 1; m >>= 1) acc += __shfl_xor(acc, m, 32);

  const float inv = 1.0f / fmaxf(sqrtf(acc), 1e-12f);

  h8 lo, hi;
#pragma unroll
  for (int i = 0; i < 8; ++i) {
    lo[i] = (_Float16)(t[i >> 2][i & 3] * inv);
    hi[i] = (_Float16)(t[2 + (i >> 2)][i & 3] * inv);
  }
  _Float16* po = xn + row * DD + lane * 16;
  *(h8*)po       = lo;
  *(h8*)(po + 8) = hi;
}

// ---------------------------------------------------------------------------
// Kernel 2: causal max cosine similarity via v_wmma_f32_16x16x32_f16.
// Wave w -> (batch b, tile-pair p, parity c). Each wave processes row tiles
// {p, 127-p} (constant total work) and the jt's of its parity.
//
// A tile (16 rows x K=512) is hoisted into 128 VGPRs of fragments and reused
// across the whole jt loop, so the hot loop fetches only B fragments
// (2 x b128 per WMMA). Two independent accumulators break the WMMA RAW chain.
// Running max initialized at -1.0 == nan_to_num(-inf) semantics for row 0.
// ---------------------------------------------------------------------------
__global__ __launch_bounds__(256) void simmax_wmma_kernel(
    const _Float16* __restrict__ xn, float* __restrict__ pmax) {
  const int lane = threadIdx.x & 31;
  const int wave = threadIdx.x >> 5;
  const int w    = blockIdx.x * 8 + wave;   // 0..1023
  const int c    = w & 1;                   // jt parity chunk
  const int pair = (w >> 1) & 63;           // 0..63
  const int b    = w >> 7;                  // 0..7

  const _Float16* xb = xn + (size_t)b * TT * DD;
  float* pout = pmax + (size_t)c * (BB * TT) + (size_t)b * TT;

  const int half16 = lane >> 4;             // 0 | 1
  const int lane15 = lane & 15;
  const int aoff   = half16 * 16;           // A-frag K sub-offset per lane half

#pragma unroll 1
  for (int sel = 0; sel < 2; ++sel) {
    const int it = sel ? (NT - 1 - pair) : pair;
    const int r0 = it * 16;

    // ---- hoist A: 16 fragments covering K = 512 (128 VGPRs, loop-invariant)
    const _Float16* pA = xb + (size_t)(r0 + lane15) * DD;
    v16h afrag[16];
#pragma unroll
    for (int kk = 0; kk < 16; ++kk) {
      const int k0 = kk * 32;
      h8 alo = *(const h8*)(pA + k0 + aoff);
      h8 ahi = *(const h8*)(pA + k0 + aoff + 8);
      afrag[kk] = __builtin_shufflevector(alo, ahi,
                    0,1,2,3,4,5,6,7,8,9,10,11,12,13,14,15);
    }

    const int myrow0 = r0 + half16 * 8;     // C VGPR v holds row myrow0+v
    float rmax[8];
#pragma unroll
    for (int v = 0; v < 8; ++v) rmax[v] = -1.0f;

    for (int jt = c; jt <= it; jt += 2) {   // wave-uniform bound: EXEC all-ones
      const int j0  = jt * 16;
      const int col = j0 + lane15;
      // B[k][n] = xn[j0+n][k]; this lane: col lane15, 16 contiguous K values
      const _Float16* pB = xb + (size_t)(j0 + lane15) * DD + half16 * 16;

      v8f c0 = {}, c1 = {};                 // two chains -> pipelined WMMAs
#pragma unroll
      for (int kk = 0; kk < 16; kk += 2) {
        v16h b0 = *(const v16h*)(pB + kk * 32);
        v16h b1 = *(const v16h*)(pB + kk * 32 + 32);
        c0 = __builtin_amdgcn_wmma_f32_16x16x32_f16(
                 false, afrag[kk],     false, b0, (short)0, c0, false, false);
        c1 = __builtin_amdgcn_wmma_f32_16x16x32_f16(
                 false, afrag[kk + 1], false, b1, (short)0, c1, false, false);
      }
      const v8f cacc = c0 + c1;

      // strictly-causal fold: only j < i participates
#pragma unroll
      for (int v = 0; v < 8; ++v) {
        const float val = (col < myrow0 + v) ? cacc[v] : -1.0f;
        rmax[v] = fmaxf(rmax[v], val);
      }
    }

    // max over the 16 columns held by this lane's 16-lane half
#pragma unroll
    for (int m = 1; m <= 8; m <<= 1) {
#pragma unroll
      for (int v = 0; v < 8; ++v)
        rmax[v] = fmaxf(rmax[v], __shfl_xor(rmax[v], m, 32));
    }
    if (lane15 == 0) {
      float* o = pout + myrow0;
#pragma unroll
      for (int v = 0; v < 8; ++v) o[v] = rmax[v];
    }
  }
}

// ---------------------------------------------------------------------------
// Kernel 3: out = gelu_tanh(x * ((1-alpha) + alpha * exp(-tau * max_sim))).
// float4-vectorized; folds the two parity partial-max planes.
// ---------------------------------------------------------------------------
__global__ __launch_bounds__(256) void scale_gelu_kernel(
    const float* __restrict__ x, const float* __restrict__ pmax,
    const float* __restrict__ log_tau, const float* __restrict__ log_blend,
    float* __restrict__ out) {
  const float tau   = __expf(log_tau[0]);
  const float alpha = 1.0f / (1.0f + __expf(-log_blend[0]));

  const size_t idx = (size_t)blockIdx.x * blockDim.x + threadIdx.x;
  const size_t e   = idx * 4;                 // 4 floats per thread
  const size_t row = e >> 9;                  // D = 512

  const float m = fmaxf(pmax[row], pmax[(size_t)BB * TT + row]);
  const float scale = (1.0f - alpha) + alpha * __expf(-tau * m);

  v4f v = *(const v4f*)(x + e);
  v4f r;
#pragma unroll
  for (int i = 0; i < 4; ++i) {
    const float z = v[i] * scale;
    const float u = 0.79788456080286535588f * (z + 0.044715f * z * z * z);
    r[i] = 0.5f * z * (1.0f + tanhf(u));
  }
  *(v4f*)(out + e) = r;
}

// ---------------------------------------------------------------------------
extern "C" void kernel_launch(void* const* d_in, const int* in_sizes, int n_in,
                              void* d_out, int out_size, void* d_ws, size_t ws_size,
                              hipStream_t stream) {
  const float* x         = (const float*)d_in[0];
  const float* log_tau   = (const float*)d_in[1];
  const float* log_blend = (const float*)d_in[2];
  float*       out       = (float*)d_out;

  // ws layout: [ xn: B*T*D f16 = 16.78 MB ][ pmax: 2*B*T f32 = 128 KB ]
  _Float16* xn  = (_Float16*)d_ws;
  float*    pmx = (float*)((char*)d_ws + (size_t)BB * TT * DD * sizeof(_Float16));

  // 1) normalize + f16 convert: 8 rows (waves) per 256-thread block
  normalize_f16_kernel<<<(BB * TT) / 8, 256, 0, stream>>>(x, xn);

  // 2) causal WMMA sim-max: 1024 balanced waves = 128 blocks of 8 waves
  simmax_wmma_kernel<<<128, 256, 0, stream>>>(xn, pmx);

  // 3) fused novelty scale + tanh-GELU: B*T*D/4 threads
  scale_gelu_kernel<<<(BB * TT * DD) / (4 * 256), 256, 0, stream>>>(
      x, pmx, log_tau, log_blend, out);
}
